// SeqCondAttention_13400297963551
// MI455X (gfx1250) — compile-verified, matrix operand order checked by reading, not run on hardware
//
#include <hip/hip_runtime.h>
#include <hip/hip_bf16.h>
#include <stdint.h>

// ---------------------------------------------------------------------------
// Problem constants (from reference)
// ---------------------------------------------------------------------------
#define BB    2
#define LL    2048
#define DD    768
#define KK    12
#define KQQ   6
#define MMM   4
#define HH    64
#define MEMC  768          // K*H
#define NCC   780          // MEM + K
#define DSWW  384
#define EXPP  192
#define LATT  192
#define BLROWS 4096        // B*L

typedef unsigned short ushort_t;
typedef __attribute__((ext_vector_type(16))) __bf16 v16bf;
typedef __attribute__((ext_vector_type(8)))  float  v8f;
typedef __attribute__((ext_vector_type(4))) unsigned int u32x4;
typedef __attribute__((ext_vector_type(8))) int         i32x8;
typedef __attribute__((ext_vector_type(4))) int         i32x4;

#if defined(__has_builtin)
#  if __has_builtin(__builtin_amdgcn_tensor_load_to_lds)
#    define HAVE_TDM 1
#  else
#    define HAVE_TDM 0
#  endif
#else
#  define HAVE_TDM 0
#endif

// ---------------------------------------------------------------------------
// bf16 helpers (storage = ushort, RNE conversion)
// ---------------------------------------------------------------------------
__device__ __forceinline__ ushort_t f2bf(float f) {
  uint32_t u = __builtin_bit_cast(uint32_t, f);
  u += 0x7fffu + ((u >> 16) & 1u);
  return (ushort_t)(u >> 16);
}
__device__ __forceinline__ float bf2f(ushort_t h) {
  uint32_t u = ((uint32_t)h) << 16;
  return __builtin_bit_cast(float, u);
}
__device__ __forceinline__ float softplusf(float x) {
  return (x > 20.f) ? x : log1pf(__expf(x));
}

// ---------------------------------------------------------------------------
// Tensor Data Mover: 2D bf16 tile (tile_d1 rows x tile_d0 elems) from global
// (row stride = stride0 elements) into LDS at lds_addr, rows packed tight.
// D# layout per CDNA5 ISA. OOB rows/cols read as zero.
// ---------------------------------------------------------------------------
__device__ __forceinline__ void tdm_load_2d_bf16(unsigned lds_addr,
                                                 const ushort_t* gptr,
                                                 unsigned tile_d0, unsigned tile_d1,
                                                 unsigned tensor_d0, unsigned tensor_d1,
                                                 unsigned stride0) {
#if HAVE_TDM
  unsigned long long ga = (unsigned long long)(uintptr_t)gptr;
  u32x4 g0;
  g0[0] = 1u;                                        // count=1, user mode
  g0[1] = lds_addr;                                  // LDS byte address
  g0[2] = (unsigned)(ga & 0xFFFFFFFFull);            // global_addr[31:0]
  g0[3] = (unsigned)((ga >> 32) & 0x01FFFFFFull)     // global_addr[56:32]
          | (2u << 30);                              // type=2 ("image")
  unsigned long long s0 = (unsigned long long)stride0;
  i32x8 g1;
  g1[0] = (int)(1u << 16);                           // data_size=1 (2 bytes)
  g1[1] = (int)((tensor_d0 & 0xFFFFu) << 16);        // tensor_dim0[15:0]
  g1[2] = (int)(((tensor_d0 >> 16) & 0xFFFFu) |      // tensor_dim0[31:16]
                ((tensor_d1 & 0xFFFFu) << 16));      // tensor_dim1[15:0]
  g1[3] = (int)(((tensor_d1 >> 16) & 0xFFFFu) |      // tensor_dim1[31:16]
                ((tile_d0 & 0xFFFFu) << 16));        // tile_dim0
  g1[4] = (int)(tile_d1 & 0xFFFFu);                  // tile_dim1 (tile_dim2=0)
  g1[5] = (int)(s0 & 0xFFFFFFFFull);                 // tensor_dim0_stride lo
  g1[6] = (int)((s0 >> 32) & 0xFFFFull);             // stride hi (dim1 stride=0)
  g1[7] = 0;
  i32x4 z4 = {0, 0, 0, 0};
#if __clang_major__ >= 23
  i32x8 z8 = {0, 0, 0, 0, 0, 0, 0, 0};
  __builtin_amdgcn_tensor_load_to_lds(g0, g1, z4, z4, z8, 0);
#else
  __builtin_amdgcn_tensor_load_to_lds(g0, g1, z4, z4, 0);
#endif
#else
  (void)lds_addr; (void)gptr; (void)tile_d0; (void)tile_d1;
  (void)tensor_d0; (void)tensor_d1; (void)stride0;
#endif
}

// ---------------------------------------------------------------------------
// f32 -> bf16 conversion kernels (plain and transposing)
// ---------------------------------------------------------------------------
__global__ void cvt_bf16_kernel(const float* __restrict__ in,
                                ushort_t* __restrict__ out, int n) {
  int i = blockIdx.x * blockDim.x + threadIdx.x;
  if (i < n) out[i] = f2bf(in[i]);
}

// src (Kd x N) row-major f32  ->  dst (N x Kd) row-major bf16
__global__ void cvt_bf16_t_kernel(const float* __restrict__ in,
                                  ushort_t* __restrict__ out, int Kd, int N) {
  int i = blockIdx.x * blockDim.x + threadIdx.x;
  if (i >= Kd * N) return;
  int kd = i % Kd;
  int n  = i / Kd;
  out[(size_t)n * Kd + kd] = f2bf(in[(size_t)kd * N + n]);
}

// W_re/W_im (K,H,DSW) -> bf16 transposed per-k B: (K, DSW, 128)
// col 0..63 = W_re[k,h,n], col 64..127 = W_im[k,h,n]
__global__ void pack_wreim_kernel(const float* __restrict__ wre,
                                  const float* __restrict__ wim,
                                  ushort_t* __restrict__ dst) {
  int idx = blockIdx.x * blockDim.x + threadIdx.x;
  if (idx >= KK * HH * DSWW) return;
  int n = idx % DSWW;
  int h = (idx / DSWW) % HH;
  int k = idx / (DSWW * HH);
  size_t kb = (size_t)k * DSWW * 128 + (size_t)n * 128;
  dst[kb + h]      = f2bf(wre[idx]);
  dst[kb + 64 + h] = f2bf(wim[idx]);
}

// ---------------------------------------------------------------------------
// bf16 WMMA GEMM: C[M,N](+batch) = A[M,K] * Bt[N,K]^T
//   A row-major bf16 (lda=Kdim-ish), Bt row-major bf16 with ldb = Kdim.
//   Tile 128x64, BK=32. 4 waves; wave = 32 rows x 64 cols = 8 WMMA / K-step.
//   Tiles pulled into LDS by the Tensor Data Mover (wave 0 issues the D#,
//   s_wait_tensorcnt 0, workgroup barrier). Requires Mrows % 128 == 0 and
//   Kdim % 32 == 0 (true for every call site).
// ---------------------------------------------------------------------------
__launch_bounds__(128)
__global__ void gemm_bf16_wmma(const ushort_t* __restrict__ A,
                               const ushort_t* __restrict__ Bt,
                               void* __restrict__ Cv,
                               int Mrows, int Ncols, int Kdim,
                               int lda, int ldc,
                               long long aBatch, long long bBatch, long long cBatch,
                               int outBf16)
{
  A  += (long long)blockIdx.z * aBatch;
  Bt += (long long)blockIdx.z * bBatch;

  __shared__ __attribute__((aligned(64))) ushort_t ldsA[128][32]; // 8 KB
  __shared__ __attribute__((aligned(64))) ushort_t ldsB[64][32];  // 4 KB

  const int tid  = threadIdx.x;
  const int wave = tid >> 5;
  const int lane = tid & 31;
  const int l16  = lane & 15;
  const int kseg = lane >> 4;

  const int tileM = blockIdx.y * 128;
  const int tileN = blockIdx.x * 64;
  const int nRows = Ncols - tileN;       // valid B rows in this tile (may be <64)

  v8f acc[2][4];
#pragma unroll
  for (int mi = 0; mi < 2; ++mi)
#pragma unroll
    for (int ni = 0; ni < 4; ++ni)
#pragma unroll
      for (int j = 0; j < 8; ++j) acc[mi][ni][j] = 0.0f;

#if HAVE_TDM
  const unsigned ldsA_addr = (unsigned)(uintptr_t)(&ldsA[0][0]);
  const unsigned ldsB_addr = (unsigned)(uintptr_t)(&ldsB[0][0]);
#endif

  for (int k0 = 0; k0 < Kdim; k0 += 32) {
#if HAVE_TDM
    if (tid < 32) {   // one wave drives the TDM; EXEC is ignored by TENSOR ops
      tdm_load_2d_bf16(ldsA_addr, A + (size_t)tileM * lda + k0,
                       /*tile*/32, 128, /*tensor*/32, 128, (unsigned)lda);
      tdm_load_2d_bf16(ldsB_addr, Bt + (size_t)tileN * Kdim + k0,
                       /*tile*/32, 64, /*tensor*/32,
                       (unsigned)(nRows > 0 ? nRows : 0), (unsigned)Kdim);
      __builtin_amdgcn_s_wait_tensorcnt(0);
    }
    __syncthreads();
#else
    // fallback: vectorized cooperative loads
#pragma unroll
    for (int p = 0; p < 4; ++p) {
      int v = tid + p * 128;
      int r = v >> 2;
      int c = (v & 3) * 8;
      uint4 val = *(const uint4*)(A + (size_t)(tileM + r) * lda + k0 + c);
      *(uint4*)(&ldsA[r][c]) = val;
    }
#pragma unroll
    for (int p = 0; p < 2; ++p) {
      int v = tid + p * 128;
      int r = v >> 2;
      int c = (v & 3) * 8;
      uint4 val = {0u, 0u, 0u, 0u};
      if (r < nRows)
        val = *(const uint4*)(Bt + (size_t)(tileN + r) * Kdim + k0 + c);
      *(uint4*)(&ldsB[r][c]) = val;
    }
    if (k0 + 32 < Kdim)
      __builtin_prefetch(A + (size_t)(tileM + (tid >> 2)) * lda + k0 + 32, 0, 1);
    __syncthreads();
#endif

    // --- WMMA: wave covers rows [32*wave, 32*wave+32) x 64 cols ---
    v16bf a0 = *(const v16bf*)(&ldsA[32 * wave + l16][kseg * 16]);
    v16bf a1 = *(const v16bf*)(&ldsA[32 * wave + 16 + l16][kseg * 16]);
#pragma unroll
    for (int ni = 0; ni < 4; ++ni) {
      v16bf bv = *(const v16bf*)(&ldsB[16 * ni + l16][kseg * 16]);
      acc[0][ni] = __builtin_amdgcn_wmma_f32_16x16x32_bf16(
          false, a0, false, bv, (short)0, acc[0][ni], false, false);
      acc[1][ni] = __builtin_amdgcn_wmma_f32_16x16x32_bf16(
          false, a1, false, bv, (short)0, acc[1][ni], false, false);
    }
    __syncthreads();
  }

  // --- store C: VGPR r -> row r + 8*kseg, lane l16 -> col ---
  if (outBf16) {
    ushort_t* C = (ushort_t*)Cv + (long long)blockIdx.z * cBatch;
#pragma unroll
    for (int ni = 0; ni < 4; ++ni) {
      int col = tileN + ni * 16 + l16;
      if (col >= Ncols) continue;
#pragma unroll
      for (int mi = 0; mi < 2; ++mi)
#pragma unroll
        for (int r = 0; r < 8; ++r) {
          int row = tileM + 32 * wave + 16 * mi + r + 8 * kseg;
          C[(size_t)row * ldc + col] = f2bf(acc[mi][ni][r]);
        }
    }
  } else {
    float* C = (float*)Cv + (long long)blockIdx.z * cBatch;
#pragma unroll
    for (int ni = 0; ni < 4; ++ni) {
      int col = tileN + ni * 16 + l16;
      if (col >= Ncols) continue;
#pragma unroll
      for (int mi = 0; mi < 2; ++mi)
#pragma unroll
        for (int r = 0; r < 8; ++r) {
          int row = tileM + 32 * wave + 16 * mi + r + 8 * kseg;
          C[(size_t)row * ldc + col] = acc[mi][ni][r];
        }
    }
  }
}

// ---------------------------------------------------------------------------
// gate[b,l,k] = sigmoid(x[b,l,:] . gate_W[:,k] + gate_b[k])
// ---------------------------------------------------------------------------
__global__ void gate_kernel(const float* __restrict__ x,
                            const float* __restrict__ gW,
                            const float* __restrict__ gb,
                            float* __restrict__ gate) {
  int idx = blockIdx.x * blockDim.x + threadIdx.x;
  if (idx >= BLROWS * KK) return;
  int k  = idx % KK;
  int bl = idx / KK;
  const float* xr = x + (size_t)bl * DD;
  float acc = 0.f;
  for (int d = 0; d < DD; ++d) acc += xr[d] * gW[d * KK + k];
  acc += gb[k];
  gate[idx] = 1.f / (1.f + __expf(-acc));
}

// ---------------------------------------------------------------------------
// Fused causal-conv + decay/score weighting + running-sum scan + q-contraction.
// One thread per (b,k,h); sequential over L. Writes bf16 A-matrix for the
// spectral GEMM: aspec (B, L, K, 128) = [out_re*norm | out_im*norm].
// ---------------------------------------------------------------------------
__launch_bounds__(64)
__global__ void scan_attn_kernel(const float* __restrict__ z0,          // (B,L,780)
                                 const ushort_t* __restrict__ qbf,      // (B,L,3072)
                                 const float* __restrict__ conv_k,      // (4,1,780)
                                 const float* __restrict__ theta_d_raw, // (K,H,M)
                                 const float* __restrict__ decay_slopes,// (K)
                                 const float* __restrict__ score_scale, // (K)
                                 const float* __restrict__ tanh_scale,  // (K)
                                 const float* __restrict__ norm_scale,  // (K,H)
                                 ushort_t* __restrict__ aspec)          // (B,L,K,128)
{
  const int b  = blockIdx.x / KK;
  const int k  = blockIdx.x % KK;
  const int h  = threadIdx.x;
  const int kq = k >> 1;   // NREP = 2

  float td[4], cum[4], theta[4];
  float run = 0.f;
#pragma unroll
  for (int m = 0; m < 4; ++m) {
    td[m] = softplusf(theta_d_raw[(k * HH + h) * MMM + m]) + 1e-4f;
    run += td[m];
    cum[m] = run;
  }
  const float total = cum[3];
  const float rng = 3.0f - 0.001f;
#pragma unroll
  for (int m = 0; m < 4; ++m) theta[m] = 0.001f + cum[m] / total * rng;
  float dth[3];
#pragma unroll
  for (int j = 0; j < 3; ++j) dth[j] = (cum[j + 1] - cum[j]) * (rng / total);
  const float wint[4] = {0.5f * dth[0], 0.5f * (dth[0] + dth[1]),
                         0.5f * (dth[1] + dth[2]), 0.5f * dth[2]};

  const float slope = softplusf(decay_slopes[k]);
  const float ssc = score_scale[k];
  const float tsc = tanh_scale[k];
  const float nsc = norm_scale[k * HH + h];

  const int c0 = k * HH + h;   // k_val channel
  const int c1 = MEMC + k;     // s_raw channel
  float w0[4], w1[4];
#pragma unroll
  for (int j = 0; j < 4; ++j) {
    w0[j] = conv_k[j * NCC + c0];
    w1[j] = conv_k[j * NCC + c1];
  }

  float win0[4] = {0.f, 0.f, 0.f, 0.f};
  float win1[4] = {0.f, 0.f, 0.f, 0.f};
  float den = 0.f;
  float nre[4] = {0.f, 0.f, 0.f, 0.f};
  float nim[4] = {0.f, 0.f, 0.f, 0.f};

  const float*    zrow = z0  + (size_t)b * LL * NCC;
  const ushort_t* qrow = qbf + (size_t)b * LL * (KQQ * HH * MMM * 2);
  ushort_t*       arow = aspec + (size_t)b * LL * (KK * 128);

  for (int l = 0; l < LL; ++l) {
    const float zv0 = zrow[(size_t)l * NCC + c0];
    const float zv1 = zrow[(size_t)l * NCC + c1];
    win0[0] = win0[1]; win0[1] = win0[2]; win0[2] = win0[3]; win0[3] = zv0;
    win1[0] = win1[1]; win1[1] = win1[2]; win1[2] = win1[3]; win1[3] = zv1;
    const float kval = w0[0]*win0[0] + w0[1]*win0[1] + w0[2]*win0[2] + w0[3]*win0[3];
    const float sraw = w1[0]*win1[0] + w1[1]*win1[1] + w1[2]*win1[2] + w1[3]*win1[3];

    const float logp = fminf(fmaxf(ssc * sraw, -20.f), 20.f);
    const float dist = (float)(LL - 1 - l);
    const float p = __expf(logp - slope * dist);
    den += p;
    const float phib = tanhf(tsc * kval);
    const float kvw = kval * p;
#pragma unroll
    for (int m = 0; m < 4; ++m) {
      float s, c;
      __sincosf(phib * theta[m], &s, &c);
      nre[m] += kvw * c;
      nim[m] += kvw * s;
    }
    const float inv = 1.0f / fmaxf(den, 1e-4f);

    const ushort_t* qp =
        qrow + (size_t)l * (KQQ * HH * MMM * 2) + (size_t)(kq * HH + h) * MMM * 2;
    float ore = 0.f, oim = 0.f;
#pragma unroll
    for (int m = 0; m < 4; ++m) {
      const float qr  = bf2f(qp[2 * m]);
      const float qi  = bf2f(qp[2 * m + 1]);
      const float sre = nre[m] * inv;
      const float sim = nim[m] * inv;
      ore += (sre * qr + sim * qi) * wint[m];
      oim += (sim * qr - sre * qi) * wint[m];
    }
    const size_t o = ((size_t)l * KK + k) * 128;
    arow[o + h]      = f2bf(ore * nsc);
    arow[o + 64 + h] = f2bf(oim * nsc);
  }
}

// ---------------------------------------------------------------------------
// y = (yv_spec*gate + yv_dir*hw) * silu(yg_spec*gate + yg_dir*hw) -> bf16
// ---------------------------------------------------------------------------
__global__ void combine_kernel(const ushort_t* __restrict__ yspec, // (B,L,K,384) bf16
                               const ushort_t* __restrict__ ydir,  // (B,L,K,384) bf16
                               const float* __restrict__ gate,     // (B,L,K)
                               const float* __restrict__ hw,       // (K)
                               ushort_t* __restrict__ ybf) {       // (B,L,K*192) bf16
  int idx = blockIdx.x * blockDim.x + threadIdx.x;
  if (idx >= BLROWS * KK * EXPP) return;
  int j  = idx % EXPP;
  int k  = (idx / EXPP) % KK;
  int bl = idx / (EXPP * KK);
  size_t base = (size_t)bl * (KK * DSWW) + (size_t)k * DSWW;
  float g = gate[(size_t)bl * KK + k];
  float hs = hw[k];
  float yv = bf2f(yspec[base + j]) * g        + bf2f(ydir[base + j]) * hs;
  float yg = bf2f(yspec[base + EXPP + j]) * g + bf2f(ydir[base + EXPP + j]) * hs;
  float sil = yg / (1.f + __expf(-yg));
  ybf[(size_t)bl * (KK * EXPP) + (size_t)k * EXPP + j] = f2bf(yv * sil);
}

// ---------------------------------------------------------------------------
// Host-side orchestration
// ---------------------------------------------------------------------------
extern "C" void kernel_launch(void* const* d_in, const int* in_sizes, int n_in,
                              void* d_out, int out_size, void* d_ws, size_t ws_size,
                              hipStream_t stream) {
  (void)in_sizes; (void)n_in; (void)out_size; (void)ws_size;

  const float* x            = (const float*)d_in[0];
  const float* W_mem        = (const float*)d_in[1];
  const float* conv_k       = (const float*)d_in[2];
  const float* W_q          = (const float*)d_in[3];
  const float* theta_d_raw  = (const float*)d_in[4];
  const float* decay_slopes = (const float*)d_in[5];
  const float* score_scale  = (const float*)d_in[6];
  const float* tanh_scale   = (const float*)d_in[7];
  const float* W_re         = (const float*)d_in[8];
  const float* W_im         = (const float*)d_in[9];
  const float* norm_scale   = (const float*)d_in[10];
  const float* gate_W       = (const float*)d_in[11];
  const float* gate_b       = (const float*)d_in[12];
  const float* skip_down_W  = (const float*)d_in[13];
  const float* skip_up_W    = (const float*)d_in[14];
  const float* highway_sc   = (const float*)d_in[15];
  const float* out_W        = (const float*)d_in[16];
  float* out = (float*)d_out;

  // workspace carving (256B aligned)
  char* ws = (char*)d_ws;
  size_t off = 0;
  auto alloc = [&](size_t bytes) -> void* {
    void* p = ws + off;
    off += (bytes + 255) & ~(size_t)255;
    return p;
  };
  ushort_t* xbf      = (ushort_t*)alloc((size_t)BLROWS * DD * 2);
  ushort_t* wmem_bt  = (ushort_t*)alloc((size_t)NCC * DD * 2);          // (780,768)
  ushort_t* wq_bt    = (ushort_t*)alloc((size_t)3072 * DD * 2);         // (3072,768)
  ushort_t* sd_bt    = (ushort_t*)alloc((size_t)LATT * DD * 2);         // (192,768)
  ushort_t* su_bt    = (ushort_t*)alloc((size_t)(KK * DSWW) * LATT * 2);// (4608,192)
  ushort_t* wreim_bt = (ushort_t*)alloc((size_t)KK * DSWW * 128 * 2);   // (12,384,128)
  ushort_t* outw_bt  = (ushort_t*)alloc((size_t)DD * (KK * EXPP) * 2);  // (768,2304)
  float*    z0       = (float*)   alloc((size_t)BLROWS * NCC * 4);
  ushort_t* qbf      = (ushort_t*)alloc((size_t)BLROWS * 3072 * 2);
  float*    gate     = (float*)   alloc((size_t)BLROWS * KK * 4);
  ushort_t* aspec    = (ushort_t*)alloc((size_t)BLROWS * KK * 128 * 2);
  ushort_t* tbf      = (ushort_t*)alloc((size_t)BLROWS * LATT * 2);
  ushort_t* ydir     = (ushort_t*)alloc((size_t)BLROWS * KK * DSWW * 2);
  ushort_t* yspec    = (ushort_t*)alloc((size_t)BLROWS * KK * DSWW * 2);
  ushort_t* ybf      = (ushort_t*)alloc((size_t)BLROWS * KK * EXPP * 2);

  const dim3 blk128(128);
  auto cvtT = [&](const float* src, ushort_t* dst, int Kd, int N) {
    cvt_bf16_t_kernel<<<(Kd * N + 255) / 256, 256, 0, stream>>>(src, dst, Kd, N);
  };

  // 1) bf16 conversions (weights transposed to (N,K) for the GEMM B operand)
  cvt_bf16_kernel<<<(BLROWS * DD + 255) / 256, 256, 0, stream>>>(x, xbf, BLROWS * DD);
  cvtT(W_mem,       wmem_bt, DD, NCC);
  cvtT(W_q,         wq_bt,   DD, 3072);
  cvtT(skip_down_W, sd_bt,   DD, LATT);
  cvtT(skip_up_W,   su_bt,   LATT, KK * DSWW);
  cvtT(out_W,       outw_bt, KK * EXPP, DD);
  pack_wreim_kernel<<<(KK * HH * DSWW + 255) / 256, 256, 0, stream>>>(
      W_re, W_im, wreim_bt);

  // 2) z0 = x @ W_mem  (4096 x 780, K=768) -> f32
  gemm_bf16_wmma<<<dim3((NCC + 63) / 64, BLROWS / 128, 1), blk128, 0, stream>>>(
      xbf, wmem_bt, z0, BLROWS, NCC, DD, DD, NCC, 0, 0, 0, /*bf16*/0);

  // 3) q = x @ W_q  (4096 x 3072, K=768) -> bf16
  gemm_bf16_wmma<<<dim3(3072 / 64, BLROWS / 128, 1), blk128, 0, stream>>>(
      xbf, wq_bt, qbf, BLROWS, 3072, DD, DD, 3072, 0, 0, 0, 1);

  // 4) gate = sigmoid(x @ gate_W + gate_b)
  gate_kernel<<<(BLROWS * KK + 255) / 256, 256, 0, stream>>>(
      x, gate_W, gate_b, gate);

  // 5) fused conv + scan + q-contraction -> aspec bf16 (B,L,K,128)
  scan_attn_kernel<<<BB * KK, HH, 0, stream>>>(
      z0, qbf, conv_k, theta_d_raw, decay_slopes, score_scale, tanh_scale,
      norm_scale, aspec);

  // 6) yspec_k = aspec_k @ [W_re;W_im]_k  (12 batched GEMMs, 4096x384, K=128)
  gemm_bf16_wmma<<<dim3(DSWW / 64, BLROWS / 128, KK), blk128, 0, stream>>>(
      aspec, wreim_bt, yspec, BLROWS, DSWW, 128,
      KK * 128, KK * DSWW,
      /*aBatch*/128, /*bBatch*/(long long)DSWW * 128, /*cBatch*/DSWW, 1);

  // 7) t = x @ skip_down_W  (4096 x 192, K=768) -> bf16
  gemm_bf16_wmma<<<dim3(LATT / 64, BLROWS / 128, 1), blk128, 0, stream>>>(
      xbf, sd_bt, tbf, BLROWS, LATT, DD, DD, LATT, 0, 0, 0, 1);

  // 8) ydir = t @ skip_up_W  (4096 x 4608, K=192) -> bf16
  gemm_bf16_wmma<<<dim3((KK * DSWW) / 64, BLROWS / 128, 1), blk128, 0, stream>>>(
      tbf, su_bt, ydir, BLROWS, KK * DSWW, LATT, LATT, KK * DSWW, 0, 0, 0, 1);

  // 9) combine gate/highway + silu -> y bf16 (4096 x 2304)
  combine_kernel<<<(BLROWS * KK * EXPP + 255) / 256, 256, 0, stream>>>(
      yspec, ydir, gate, highway_sc, ybf);

  // 10) out = y @ out_W  (4096 x 768, K=2304) -> f32 d_out
  gemm_bf16_wmma<<<dim3(DD / 64, BLROWS / 128, 1), blk128, 0, stream>>>(
      ybf, outw_bt, out, BLROWS, DD, KK * EXPP, KK * EXPP, DD, 0, 0, 0, 0);
}
